// HOElementWiseGRU_42090679500829
// MI455X (gfx1250) — compile-verified, matrix operand order checked by reading, not run on hardware
//
#include <hip/hip_runtime.h>
#include <math.h>

// ---------------- problem constants ----------------
#define HID   512
#define CBATCH 16384
#define FF    81919
#define GG    5
#define NINC  5
#define NLAY  2
#define MT    32                      // rows per block
#define KSTEPS 16                     // 512 / 32
#define ZR_HALVES (1024*512)          // zr section: 1024 real cols x K=512
#define A_HALVES  (512*512)           // a  section:  512 real cols x K=512
#define PACK_PER_MAT ((size_t)3*ZR_HALVES + (size_t)3*A_HALVES)
#define OUT_FLOATS (2*FF)             // complex out, interleaved

typedef __attribute__((ext_vector_type(16))) __bf16 v16bf;
typedef __attribute__((ext_vector_type(8)))  float  v8f;

union BV { uint4 q[2]; v16bf v; };

__device__ __forceinline__ unsigned short f2bf(float f) {
  union { float f; unsigned u; } cv; cv.f = f;
  unsigned u = cv.u;
  return (unsigned short)((u + 0x7FFFu + ((u >> 16) & 1u)) >> 16);   // RNE
}
__device__ __forceinline__ float bf2f(unsigned short h) {
  union { unsigned u; float f; } cv; cv.u = ((unsigned)h) << 16; return cv.f;
}

// A operand (16-bit, 16x32): per lane two 16B chunks located 32B apart in K.
__device__ __forceinline__ v16bf loadA(const unsigned short* p) {
  BV b; b.q[0] = *(const uint4*)(p); b.q[1] = *(const uint4*)(p + 16); return b.v;
}
// B operand: pre-packed contiguous 32B per lane.
__device__ __forceinline__ v16bf loadB(const unsigned short* p) {
  BV b; const uint4* q = (const uint4*)p; b.q[0] = q[0]; b.q[1] = q[1]; return b.v;
}

__device__ __forceinline__ v8f wmma_bf16(v16bf a, v16bf b, v8f c) {
  return __builtin_amdgcn_wmma_f32_16x16x32_bf16(false, a, false, b, (short)0, c, false, false);
}

// ------------- kernel 0: magnitude-compressed stack -------------
__global__ void compress_stack(const float* __restrict__ x, const float* __restrict__ ex,
                               float* __restrict__ comp) {
  int t = blockIdx.x * blockDim.x + threadIdx.x;
  if (t >= 81920 * NINC) return;
  int f = t / NINC, j = t - f * NINC;
  float re = 0.f, im = 0.f;
  if (f < FF) {
    if (j == 0) { re = x[2*f]; im = x[2*f+1]; }
    else { size_t e = (size_t)(j-1)*FF + f; re = ex[2*e]; im = ex[2*e+1]; }
  }
  float mag = sqrtf(re*re + im*im);
  float s = (mag > 0.f) ? (log1pf(mag)/mag) : 0.f;
  comp[2*t] = re*s; comp[2*t+1] = im*s;
}

// ------------- kernel 1: pack complex weights into WMMA-B planes -------------
__global__ void pack_weights(const float* __restrict__ w_i, const float* __restrict__ w_h,
                             unsigned short* __restrict__ pk) {
  int t = blockIdx.x * blockDim.x + threadIdx.x;
  if (t >= NLAY * 2 * 1536 * HID) return;
  int k   = t & (HID - 1);
  int n   = (t >> 9) % 1536;
  int mat = ((t >> 9) / 1536) & 1;
  int l   = t / (HID * 1536 * 2);
  const float* src = mat ? w_h : w_i;
  size_t si = (((size_t)l*HID + k)*1536 + n)*2;
  float re = src[si], im = src[si+1];
  unsigned short* base = pk + (size_t)(l*2 + mat) * PACK_PER_MAT;
  int nl_; size_t sec; size_t planeSz;
  if (n < 1024) { nl_ = n;        sec = 0;                   planeSz = ZR_HALVES; }
  else          { nl_ = n - 1024; sec = (size_t)3*ZR_HALVES; planeSz = A_HALVES;  }
  int ct = nl_ >> 4;
  int tt = k >> 5;
  int kr = k & 31;
  int lane = (nl_ & 15) + (((kr >> 3) & 1) << 4);
  int jj   = (kr & 7) + ((kr >> 4) << 3);
  size_t pidx = ((size_t)(ct * KSTEPS + tt) * 32 + lane) * 16 + jj;
  base[sec + pidx]             = f2bf(re);
  base[sec + planeSz + pidx]   = f2bf(im);
  base[sec + 2*planeSz + pidx] = f2bf(-im);
}

// ------------- kernel 2: input projection (K=25, scalar f32) -------------
__global__ void input_proj(const float* __restrict__ comp, const float* __restrict__ w_in,
                           const float* __restrict__ b_in,
                           unsigned short* __restrict__ xr, unsigned short* __restrict__ xi) {
  int t = blockIdx.x * blockDim.x + threadIdx.x;
  if (t >= CBATCH * HID) return;
  int c = t >> 9, n = t & (HID - 1);
  float ar = b_in[2*n], ai = b_in[2*n+1];
  const float* cp = comp + (size_t)c * (NINC*GG) * 2;
  #pragma unroll
  for (int k = 0; k < NINC*GG; ++k) {
    float sr = cp[2*k], si = cp[2*k+1];
    float wr = w_in[((size_t)k*HID + n)*2], wi = w_in[((size_t)k*HID + n)*2 + 1];
    ar += sr*wr - si*wi;
    ai += sr*wi + si*wr;
  }
  xr[t] = f2bf(ar); xi[t] = f2bf(ai);
}

// ------------- kernel 3: fused complex GRU layer (WMMA bf16) -------------
// One block = 32 batch rows (2 row-blocks per wave), 16 waves.
// K-loop is manually unrolled x2 with ping-pong B buffers so the 6 global B
// loads for step t+1 are in flight while the 16 WMMAs of step t execute.
__global__ void __launch_bounds__(512)
cgru_layer(const unsigned short* __restrict__ xr, const unsigned short* __restrict__ xi,
           const float* __restrict__ h_in,
           const unsigned short* __restrict__ wi, const unsigned short* __restrict__ wh,
           const float* __restrict__ b,
           float* __restrict__ h_out,
           unsigned short* __restrict__ yr, unsigned short* __restrict__ yi) {
  __shared__ unsigned short s_xr[MT*HID], s_xi[MT*HID];
  __shared__ unsigned short s_sr[MT*HID], s_si[MT*HID];
  __shared__ unsigned short s_rr[MT*HID], s_ri[MT*HID];
  __shared__ unsigned short s_zr[MT*HID], s_zi[MT*HID];

  const int tid = threadIdx.x;
  const int c0  = blockIdx.x * MT;
  for (int idx = tid; idx < MT*HID; idx += 512) {
    int m = idx >> 9, k = idx & (HID - 1);
    size_t g = (size_t)(c0 + m) * HID + k;
    s_xr[idx] = xr[g]; s_xi[idx] = xi[g];
    s_sr[idx] = f2bf(h_in[2*g]); s_si[idx] = f2bf(h_in[2*g+1]);
  }
  __syncthreads();

  const int wave = tid >> 5, lane = tid & 31;
  const int a0 = (lane & 15) * HID + ((lane >> 4) << 3);        // row block 0
  const int a1 = a0 + 16 * HID;                                 // row block 1
  const int mbase = (lane >> 4) << 3;

  const unsigned short *wiR = wi, *wiI = wi + ZR_HALVES, *wiN = wi + 2*(size_t)ZR_HALVES;
  const unsigned short *whR = wh, *whI = wh + ZR_HALVES, *whN = wh + 2*(size_t)ZR_HALVES;

  // ---- zr gates: sigma(x@Wi_zr + s@Wh_zr + b_zr) ----
  for (int rep = 0; rep < 4; ++rep) {
    int ct = wave + rep * 16;                      // complex col-tile 0..63
    v8f cr0 = {0.f,0.f,0.f,0.f,0.f,0.f,0.f,0.f};
    v8f ci0 = cr0, cr1 = cr0, ci1 = cr0;
    size_t base = (size_t)ct * (KSTEPS*32*16) + (size_t)lane * 16;
    v16bf p0 = loadB(wiR + base), p1 = loadB(wiI + base), p2 = loadB(wiN + base);
    v16bf p3 = loadB(whR + base), p4 = loadB(whI + base), p5 = loadB(whN + base);
    #pragma unroll 1
    for (int t = 0; t < KSTEPS; t += 2) {
      size_t bo1 = base + (size_t)(t + 1) * 512;
      v16bf q0 = loadB(wiR + bo1), q1 = loadB(wiI + bo1), q2 = loadB(wiN + bo1);
      v16bf q3 = loadB(whR + bo1), q4 = loadB(whI + bo1), q5 = loadB(whN + bo1);
      {
        int ko = t * 32;
        v16bf xr0 = loadA(s_xr + a0 + ko), xi0 = loadA(s_xi + a0 + ko);
        v16bf sr0 = loadA(s_sr + a0 + ko), si0 = loadA(s_si + a0 + ko);
        v16bf xr1 = loadA(s_xr + a1 + ko), xi1 = loadA(s_xi + a1 + ko);
        v16bf sr1 = loadA(s_sr + a1 + ko), si1 = loadA(s_si + a1 + ko);
        cr0 = wmma_bf16(xr0, p0, cr0);  cr0 = wmma_bf16(xi0, p2, cr0);
        cr0 = wmma_bf16(sr0, p3, cr0);  cr0 = wmma_bf16(si0, p5, cr0);
        ci0 = wmma_bf16(xr0, p1, ci0);  ci0 = wmma_bf16(xi0, p0, ci0);
        ci0 = wmma_bf16(sr0, p4, ci0);  ci0 = wmma_bf16(si0, p3, ci0);
        cr1 = wmma_bf16(xr1, p0, cr1);  cr1 = wmma_bf16(xi1, p2, cr1);
        cr1 = wmma_bf16(sr1, p3, cr1);  cr1 = wmma_bf16(si1, p5, cr1);
        ci1 = wmma_bf16(xr1, p1, ci1);  ci1 = wmma_bf16(xi1, p0, ci1);
        ci1 = wmma_bf16(sr1, p4, ci1);  ci1 = wmma_bf16(si1, p3, ci1);
      }
      size_t bo2 = base + (size_t)((t + 2 < KSTEPS) ? t + 2 : t) * 512;
      p0 = loadB(wiR + bo2); p1 = loadB(wiI + bo2); p2 = loadB(wiN + bo2);
      p3 = loadB(whR + bo2); p4 = loadB(whI + bo2); p5 = loadB(whN + bo2);
      {
        int ko = (t + 1) * 32;
        v16bf xr0 = loadA(s_xr + a0 + ko), xi0 = loadA(s_xi + a0 + ko);
        v16bf sr0 = loadA(s_sr + a0 + ko), si0 = loadA(s_si + a0 + ko);
        v16bf xr1 = loadA(s_xr + a1 + ko), xi1 = loadA(s_xi + a1 + ko);
        v16bf sr1 = loadA(s_sr + a1 + ko), si1 = loadA(s_si + a1 + ko);
        cr0 = wmma_bf16(xr0, q0, cr0);  cr0 = wmma_bf16(xi0, q2, cr0);
        cr0 = wmma_bf16(sr0, q3, cr0);  cr0 = wmma_bf16(si0, q5, cr0);
        ci0 = wmma_bf16(xr0, q1, ci0);  ci0 = wmma_bf16(xi0, q0, ci0);
        ci0 = wmma_bf16(sr0, q4, ci0);  ci0 = wmma_bf16(si0, q3, ci0);
        cr1 = wmma_bf16(xr1, q0, cr1);  cr1 = wmma_bf16(xi1, q2, cr1);
        cr1 = wmma_bf16(sr1, q3, cr1);  cr1 = wmma_bf16(si1, q5, cr1);
        ci1 = wmma_bf16(xr1, q1, ci1);  ci1 = wmma_bf16(xi1, q0, ci1);
        ci1 = wmma_bf16(sr1, q4, ci1);  ci1 = wmma_bf16(si1, q3, ci1);
      }
    }
    int col = ct * 16 + (lane & 15);
    float bre = b[2*col], bim = b[2*col+1];
    #pragma unroll
    for (int rb = 0; rb < 2; ++rb) {
      const v8f& cr = rb ? cr1 : cr0;
      const v8f& ci = rb ? ci1 : ci0;
      #pragma unroll
      for (int j = 0; j < 8; ++j) {
        int m = rb*16 + mbase + j;
        float zre = 1.f / (1.f + __expf(-(cr[j] + bre)));
        float zim = 1.f / (1.f + __expf(-(ci[j] + bim)));
        if (col < HID) { s_zr[m*HID + col] = f2bf(zre); s_zi[m*HID + col] = f2bf(zim); }
        else {
          int hd = col - HID;
          float sr = bf2f(s_sr[m*HID + hd]), si = bf2f(s_si[m*HID + hd]);
          s_rr[m*HID + hd] = f2bf(zre*sr - zim*si);   // r * state (complex)
          s_ri[m*HID + hd] = f2bf(zre*si + zim*sr);
        }
      }
    }
  }
  __syncthreads();

  // ---- a gate: tanh(x@Wi_a + (r*s)@Wh_a + b_a); state update ----
  const unsigned short *waR = wi + 3*(size_t)ZR_HALVES, *waI = waR + A_HALVES, *waN = waR + 2*(size_t)A_HALVES;
  const unsigned short *hbR = wh + 3*(size_t)ZR_HALVES, *hbI = hbR + A_HALVES, *hbN = hbR + 2*(size_t)A_HALVES;
  for (int rep = 0; rep < 2; ++rep) {
    int ct = wave + rep * 16;                      // complex col-tile 0..31
    v8f cr0 = {0.f,0.f,0.f,0.f,0.f,0.f,0.f,0.f};
    v8f ci0 = cr0, cr1 = cr0, ci1 = cr0;
    size_t base = (size_t)ct * (KSTEPS*32*16) + (size_t)lane * 16;
    v16bf p0 = loadB(waR + base), p1 = loadB(waI + base), p2 = loadB(waN + base);
    v16bf p3 = loadB(hbR + base), p4 = loadB(hbI + base), p5 = loadB(hbN + base);
    #pragma unroll 1
    for (int t = 0; t < KSTEPS; t += 2) {
      size_t bo1 = base + (size_t)(t + 1) * 512;
      v16bf q0 = loadB(waR + bo1), q1 = loadB(waI + bo1), q2 = loadB(waN + bo1);
      v16bf q3 = loadB(hbR + bo1), q4 = loadB(hbI + bo1), q5 = loadB(hbN + bo1);
      {
        int ko = t * 32;
        v16bf xr0 = loadA(s_xr + a0 + ko), xi0 = loadA(s_xi + a0 + ko);
        v16bf rr0 = loadA(s_rr + a0 + ko), ri0 = loadA(s_ri + a0 + ko);
        v16bf xr1 = loadA(s_xr + a1 + ko), xi1 = loadA(s_xi + a1 + ko);
        v16bf rr1 = loadA(s_rr + a1 + ko), ri1 = loadA(s_ri + a1 + ko);
        cr0 = wmma_bf16(xr0, p0, cr0);  cr0 = wmma_bf16(xi0, p2, cr0);
        cr0 = wmma_bf16(rr0, p3, cr0);  cr0 = wmma_bf16(ri0, p5, cr0);
        ci0 = wmma_bf16(xr0, p1, ci0);  ci0 = wmma_bf16(xi0, p0, ci0);
        ci0 = wmma_bf16(rr0, p4, ci0);  ci0 = wmma_bf16(ri0, p3, ci0);
        cr1 = wmma_bf16(xr1, p0, cr1);  cr1 = wmma_bf16(xi1, p2, cr1);
        cr1 = wmma_bf16(rr1, p3, cr1);  cr1 = wmma_bf16(ri1, p5, cr1);
        ci1 = wmma_bf16(xr1, p1, ci1);  ci1 = wmma_bf16(xi1, p0, ci1);
        ci1 = wmma_bf16(rr1, p4, ci1);  ci1 = wmma_bf16(ri1, p3, ci1);
      }
      size_t bo2 = base + (size_t)((t + 2 < KSTEPS) ? t + 2 : t) * 512;
      p0 = loadB(waR + bo2); p1 = loadB(waI + bo2); p2 = loadB(waN + bo2);
      p3 = loadB(hbR + bo2); p4 = loadB(hbI + bo2); p5 = loadB(hbN + bo2);
      {
        int ko = (t + 1) * 32;
        v16bf xr0 = loadA(s_xr + a0 + ko), xi0 = loadA(s_xi + a0 + ko);
        v16bf rr0 = loadA(s_rr + a0 + ko), ri0 = loadA(s_ri + a0 + ko);
        v16bf xr1 = loadA(s_xr + a1 + ko), xi1 = loadA(s_xi + a1 + ko);
        v16bf rr1 = loadA(s_rr + a1 + ko), ri1 = loadA(s_ri + a1 + ko);
        cr0 = wmma_bf16(xr0, q0, cr0);  cr0 = wmma_bf16(xi0, q2, cr0);
        cr0 = wmma_bf16(rr0, q3, cr0);  cr0 = wmma_bf16(ri0, q5, cr0);
        ci0 = wmma_bf16(xr0, q1, ci0);  ci0 = wmma_bf16(xi0, q0, ci0);
        ci0 = wmma_bf16(rr0, q4, ci0);  ci0 = wmma_bf16(ri0, q3, ci0);
        cr1 = wmma_bf16(xr1, q0, cr1);  cr1 = wmma_bf16(xi1, q2, cr1);
        cr1 = wmma_bf16(rr1, q3, cr1);  cr1 = wmma_bf16(ri1, q5, cr1);
        ci1 = wmma_bf16(xr1, q1, ci1);  ci1 = wmma_bf16(xi1, q0, ci1);
        ci1 = wmma_bf16(rr1, q4, ci1);  ci1 = wmma_bf16(ri1, q3, ci1);
      }
    }
    int col = ct * 16 + (lane & 15);
    float bre = b[2*(1024 + col)], bim = b[2*(1024 + col) + 1];
    #pragma unroll
    for (int rb = 0; rb < 2; ++rb) {
      const v8f& cr = rb ? cr1 : cr0;
      const v8f& ci = rb ? ci1 : ci0;
      #pragma unroll
      for (int j = 0; j < 8; ++j) {
        int m = rb*16 + mbase + j;
        float ag_r = tanhf(cr[j] + bre);
        float ag_i = tanhf(ci[j] + bim);
        float zre = bf2f(s_zr[m*HID + col]), zim = bf2f(s_zi[m*HID + col]);
        float sr = bf2f(s_sr[m*HID + col]), si = bf2f(s_si[m*HID + col]);
        // new = (1-z)*s + z*a  (complex)
        float nr = (1.f - zre)*sr + zim*si + zre*ag_r - zim*ag_i;
        float ni = (1.f - zre)*si - zim*sr + zre*ag_i + zim*ag_r;
        size_t g = (size_t)(c0 + m) * HID + col;
        h_out[2*g] = nr; h_out[2*g+1] = ni;
        yr[g] = f2bf(nr); yi[g] = f2bf(ni);
      }
    }
  }
}

// ------------- kernel 4: output projection (N=5, scalar f32) -------------
__global__ void output_proj(const float* __restrict__ h2, const float* __restrict__ w_out,
                            const float* __restrict__ b_out, float* __restrict__ out) {
  int t = blockIdx.x * blockDim.x + threadIdx.x;
  if (t >= FF) return;
  int c = t / GG, g = t - c * GG;
  float ar = b_out[2*g], ai = b_out[2*g+1];
  const float* hr = h2 + (size_t)c * HID * 2;
  for (int n = 0; n < HID; ++n) {
    float sr = hr[2*n], si = hr[2*n+1];
    float wr = w_out[(n*GG + g)*2], wi = w_out[(n*GG + g)*2 + 1];
    ar += sr*wr - si*wi;
    ai += sr*wi + si*wr;
  }
  out[2*t]   = -0.01f * ar;
  out[2*t+1] = -0.01f * ai;
}

extern "C" void kernel_launch(void* const* d_in, const int* in_sizes, int n_in,
                              void* d_out, int out_size, void* d_ws, size_t ws_size,
                              hipStream_t stream) {
  (void)in_sizes; (void)n_in; (void)out_size; (void)ws_size;
  const float* x     = (const float*)d_in[0];
  const float* ex    = (const float*)d_in[1];
  const float* h     = (const float*)d_in[2];
  const float* w_in  = (const float*)d_in[3];
  const float* b_in  = (const float*)d_in[4];
  const float* w_i   = (const float*)d_in[5];
  const float* w_h   = (const float*)d_in[6];
  const float* b_rnn = (const float*)d_in[7];
  const float* w_out = (const float*)d_in[8];
  const float* b_out = (const float*)d_in[9];
  float* out = (float*)d_out;

  // workspace layout (~90 MB): comp | x0 planes | x1 planes | packed weights
  char* ws = (char*)d_ws;
  float* comp = (float*)ws;
  size_t off = (size_t)81920 * NINC * 2 * sizeof(float);
  unsigned short* x0r = (unsigned short*)(ws + off); off += (size_t)CBATCH*HID*2;
  unsigned short* x0i = (unsigned short*)(ws + off); off += (size_t)CBATCH*HID*2;
  unsigned short* x1r = (unsigned short*)(ws + off); off += (size_t)CBATCH*HID*2;
  unsigned short* x1i = (unsigned short*)(ws + off); off += (size_t)CBATCH*HID*2;
  unsigned short* pk  = (unsigned short*)(ws + off);

  float* hout0 = out + OUT_FLOATS;
  float* hout1 = out + OUT_FLOATS + (size_t)CBATCH*HID*2;

  compress_stack<<<(81920*NINC + 255)/256, 256, 0, stream>>>(x, ex, comp);
  pack_weights<<<(NLAY*2*1536*HID + 255)/256, 256, 0, stream>>>(w_i, w_h, pk);
  input_proj<<<(CBATCH*HID + 255)/256, 256, 0, stream>>>(comp, w_in, b_in, x0r, x0i);

  // layer 0
  cgru_layer<<<CBATCH/MT, 512, 0, stream>>>(
      x0r, x0i, h,
      pk + (size_t)0*PACK_PER_MAT,           // w_i[0]
      pk + (size_t)1*PACK_PER_MAT,           // w_h[0]
      b_rnn,
      hout0, x1r, x1i);
  // layer 1 (bf16 planes of its output overwrite x0 planes; unused after)
  cgru_layer<<<CBATCH/MT, 512, 0, stream>>>(
      x1r, x1i, h + (size_t)CBATCH*HID*2,
      pk + (size_t)2*PACK_PER_MAT,           // w_i[1]
      pk + (size_t)3*PACK_PER_MAT,           // w_h[1]
      b_rnn + 1536*2,
      hout1, x0r, x0i);

  output_proj<<<(FF + 255)/256, 256, 0, stream>>>(hout1, w_out, b_out, out);
}
